// Model_61065845015380
// MI455X (gfx1250) — matrix operand order repaired and checked by reading, not compiled, not run on hardware
//
#include <hip/hip_runtime.h>
#include <hip/hip_bf16.h>
#include <math.h>

typedef __attribute__((ext_vector_type(16))) _Float16 v16h;
typedef __attribute__((ext_vector_type(8)))  float    v8f;

#define N_FREQ   50
#define ENC_DIM  200   // 7 K-chunks of 32 (padded to 224)
#define EMB      10
#define KCHUNKS  7
// log2(1000) * 2 / 50
#define FREQ_LOG2_STEP 0.3986313713864835f
#define INV2PI   0.15915494309189535f
#define LOG2E    1.4426950408889634f

static __device__ __forceinline__ float fast_sigmoid(float x) {
    // 1 / (1 + exp(-x)) == rcp(1 + exp2(-x*log2e)) : v_exp + v_add + v_rcp
    return __builtin_amdgcn_rcpf(1.0f + __builtin_amdgcn_exp2f(-x * LOG2E));
}

// ---------------------------------------------------------------------------
// Kernel 1: sine-encode -> WMMA layer1 -> sigmoid -> WMMA layer2 -> atomic
//           segment-sum into pooled[segments][10].
// One wave = one tile of 16 fragments. D = Aᵀ layout: D[M=embdim, N=frag].
// All math is unconditional (garbage columns never reach memory); the only
// divergent code is the final atomic guard.
// ---------------------------------------------------------------------------
__global__ void __launch_bounds__(256)
frag_embed_pool_kernel(const float* __restrict__ coords,
                       const int*   __restrict__ segIx,
                       const float* __restrict__ W1,   // [200,10]
                       const float* __restrict__ b1,   // [10]
                       const float* __restrict__ W2,   // [10,10]
                       const float* __restrict__ b2,   // [10]
                       float*       __restrict__ pooled, // [segments,10]
                       int nfrag, int ntiles)
{
    const int lane    = threadIdx.x & 31;
    const int l       = lane & 15;     // A: M index (emb dim); B: fragment slot
    const int halfw   = lane >> 4;     // lane half selects K sub-range
    const int halfw16 = halfw << 4;
    const float lowmask = (halfw == 0) ? 1.0f : 0.0f;
    const int wave    = blockIdx.x * (blockDim.x >> 5) + (threadIdx.x >> 5);
    const int nwaves  = gridDim.x * (blockDim.x >> 5);

    // ---- A1 = W1ᵀ in WMMA 16-bit A layout (16x32 per chunk) ---------------
    // slot s = vgpr*2 + hi16 ; lanes 0-15: K = 0..15, lanes 16-31: K = 16..31
    v16h a1[KCHUNKS];
    #pragma unroll
    for (int cc = 0; cc < KCHUNKS; ++cc) {
        #pragma unroll
        for (int s = 0; s < 16; ++s) {
            const int v  = s >> 1, h16 = s & 1;
            const int kl = 2 * v + h16 + halfw16;
            const int k  = cc * 32 + kl;
            float w = 0.0f;
            if (k < ENC_DIM && l < EMB) w = W1[k * EMB + l];
            a1[cc][s] = (_Float16)w;
        }
    }
    // ---- A2 = W2ᵀ (K = 10, rest zero) -------------------------------------
    v16h a2;
    #pragma unroll
    for (int s = 0; s < 16; ++s) {
        const int v  = s >> 1, h16 = s & 1;
        const int kl = 2 * v + h16 + halfw16;
        float w = 0.0f;
        if (kl < EMB && l < EMB) w = W2[kl * EMB + l];
        a2[s] = (_Float16)w;
    }
    // ---- Per-lane biases (C/D layout: vgpr r -> M = r + 8*half) -----------
    float b1v[8], b2v[8];
    #pragma unroll
    for (int r = 0; r < 8; ++r) {
        const int j = r + 8 * halfw;
        b1v[r] = (j < EMB) ? b1[j] : 0.0f;
        b2v[r] = (j < EMB) ? b2[j] : 0.0f;
    }
    // ---- Tile-invariant per-lane frequencies, pre-scaled by 1/2pi ---------
    // pair p covers B slots (2p, 2p+1) = (sin, cos) of one phase.
    float frq[KCHUNKS * 8];
    #pragma unroll
    for (int cc = 0; cc < KCHUNKS; ++cc) {
        #pragma unroll
        for (int p = 0; p < 8; ++p) {
            const int k2 = cc * 32 + halfw16 + 2 * p;  // even k
            const int j  = k2 % 100;
            const int t  = j >> 1;                     // 0..49 (junk ok if padded)
            frq[cc * 8 + p] =
                __builtin_amdgcn_exp2f(-FREQ_LOG2_STEP * (float)(t + 1)) * INV2PI;
        }
    }

    for (int tile = wave; tile < ntiles; tile += nwaves) {
        const int frag  = tile * 16 + l;
        const bool valid = frag < nfrag;
        const int fcl   = valid ? frag : (nfrag - 1);  // clamped, loads safe
        const float cx  = coords[2 * fcl];
        const float cy  = coords[2 * fcl + 1];
        const int  seg  = segIx[fcl];

        // ---- layer 1: D1 = W1ᵀ(16x224) x encᵀ(224x16), K-chunks of 32 ----
        v8f acc = {};
        #pragma unroll
        for (int cc = 0; cc < KCHUNKS; ++cc) {
            v16h bm;
            #pragma unroll
            for (int p = 0; p < 8; ++p) {
                const int klo = cc * 32 + 2 * p;       // compile-time part
                const float co = ((klo + halfw16) >= 100) ? cy : cx;
                const float ph =
                    __builtin_amdgcn_fractf(co * frq[cc * 8 + p]);
                float s0 = __builtin_amdgcn_sinf(ph);  // sin(2*pi*ph)
                float s1 = __builtin_amdgcn_cosf(ph);  // shift pi/2 partner
                if (cc == KCHUNKS - 1) {               // K padding 200..223
                    if (p < 4) { s0 *= lowmask; s1 *= lowmask; }
                    else       { s0 = 0.0f;     s1 = 0.0f;     }
                }
                bm[2 * p]     = (_Float16)s0;
                bm[2 * p + 1] = (_Float16)s1;
            }
            acc = __builtin_amdgcn_wmma_f32_16x16x32_f16(
                      false, a1[cc], false, bm, (short)0, acc, false, false);
        }

        // ---- bias + sigmoid ----------------------------------------------
        float hs[8], ho[8];
        #pragma unroll
        for (int r = 0; r < 8; ++r)
            hs[r] = fast_sigmoid(acc[r] + b1v[r]);
        // lane f needs dims 8..15 which live in lane f+16 (single shuffle)
        #pragma unroll
        for (int r = 0; r < 8; ++r) ho[r] = __shfl_xor(hs[r], 16, 32);

        // ---- layer 2: D2 = W2ᵀ(16x32) x hᵀ(32x16) ------------------------
        // B: lanes 0-15 hold K=0..15, lanes 16-31 (K=16..31) are zero.
        v16h bm2;
        #pragma unroll
        for (int s = 0; s < 16; ++s) {
            const float vv = ((s < 8) ? hs[s] : ho[s - 8]) * lowmask;
            bm2[s] = (_Float16)vv;
        }
        v8f z = {};
        v8f d2 = __builtin_amdgcn_wmma_f32_16x16x32_f16(
                     false, a2, false, bm2, (short)0, z, false, false);

        // ---- segment sum: atomics into pooled[seg][j'] -------------------
        if (valid) {
            float* base = pooled + (long)seg * EMB;
            if (halfw == 0) {
                #pragma unroll
                for (int r = 0; r < 8; ++r)
                    atomicAdd(base + r, d2[r] + b2v[r]);       // dims 0..7
            } else {
                atomicAdd(base + 8, d2[0] + b2v[0]);           // dim 8
                atomicAdd(base + 9, d2[1] + b2v[1]);           // dim 9
            }
        }
    }
}

// ---------------------------------------------------------------------------
// Kernel 2: per-segment MLP  sigmoid(p@We1+be1)@We2+be2  (55 MFLOP, BW-bound)
// ---------------------------------------------------------------------------
__global__ void __launch_bounds__(256)
expr_kernel(const float* __restrict__ pooled,
            const float* __restrict__ We1, const float* __restrict__ be1,
            const float* __restrict__ We2, const float* __restrict__ be2,
            float* __restrict__ out, int nseg)
{
    const int i = blockIdx.x * blockDim.x + threadIdx.x;
    if (i >= nseg) return;
    float p[EMB];
    #pragma unroll
    for (int n = 0; n < EMB; ++n) p[n] = pooled[(long)i * EMB + n];
    float o = be2[0];
    #pragma unroll
    for (int m = 0; m < EMB; ++m) {
        float x = be1[m];
        #pragma unroll
        for (int n = 0; n < EMB; ++n) x += p[n] * We1[n * EMB + m];
        o += fast_sigmoid(x) * We2[m];
    }
    out[i] = o;
}

extern "C" void kernel_launch(void* const* d_in, const int* in_sizes, int n_in,
                              void* d_out, int out_size, void* d_ws, size_t ws_size,
                              hipStream_t stream)
{
    const float* coords = (const float*)d_in[0];
    const int*   segIx  = (const int*)  d_in[1];
    // d_in[2]=cell_n, d_in[3]=gene_n (device scalars; segments == out_size)
    const float* W1  = (const float*)d_in[4];
    const float* b1  = (const float*)d_in[5];
    const float* W2  = (const float*)d_in[6];
    const float* b2  = (const float*)d_in[7];
    const float* We1 = (const float*)d_in[8];
    const float* be1 = (const float*)d_in[9];
    const float* We2 = (const float*)d_in[10];
    const float* be2 = (const float*)d_in[11];

    const int nfrag  = in_sizes[0] / 2;
    const int ntiles = (nfrag + 15) / 16;
    const int nseg   = out_size;

    float* pooled = (float*)d_ws;   // [nseg,10] f32 accumulator
    hipMemsetAsync(pooled, 0, (size_t)nseg * EMB * sizeof(float), stream);

    int blocks = (ntiles + 7) / 8;            // 8 wave32 tiles per block
    if (blocks > 2048) blocks = 2048;         // ~4 tiles/wave amortizes setup
    frag_embed_pool_kernel<<<blocks, 256, 0, stream>>>(
        coords, segIx, W1, b1, W2, b2, pooled, nfrag, ntiles);

    expr_kernel<<<(nseg + 255) / 256, 256, 0, stream>>>(
        pooled, We1, be1, We2, be2, (float*)d_out, nseg);
}
